// LlamaAttention_23106924052755
// MI455X (gfx1250) — matrix operand order repaired and checked by reading, not compiled, not run on hardware
//
#include <hip/hip_runtime.h>

// ---------------------------------------------------------------------------
// CDNA5 / gfx1250 Llama attention block, bf16 WMMA pipeline.
// wave32; V_WMMA_F32_16X16X32_BF16 for all matmuls; async LDS staging for the
// shared GEMM B operand (GLOBAL_LOAD_ASYNC_TO_LDS_B128 + s_wait_asynccnt).
// ---------------------------------------------------------------------------

typedef __attribute__((ext_vector_type(16))) __bf16 v16bf;
typedef __attribute__((ext_vector_type(8)))  __bf16 v8bf;
typedef __attribute__((ext_vector_type(8)))  float  v8f;

// Exact parameter type of the async-LDS builtins (from hipcc diagnostic):
// 'vector_size(4*sizeof(int)) int' pointers in AS1 (global) / AS3 (LDS).
typedef int v4i_vs __attribute__((vector_size(16)));
typedef __attribute__((address_space(1))) v4i_vs* as1_v4i_p;
typedef __attribute__((address_space(3))) v4i_vs* as3_v4i_p;
typedef __attribute__((address_space(1))) void*   as1_void_p;
typedef __attribute__((address_space(3))) void*   as3_void_p;

#define S_LEN 2048
#define D_MODEL 2048
#define N_HEADS 32
#define N_KV 8
#define HEAD_DIM 64

#if defined(__gfx1250__) && __has_builtin(__builtin_amdgcn_global_load_async_to_lds_b128)
#define USE_ASYNC_LDS 1
#else
#define USE_ASYNC_LDS 0
#endif

__device__ inline v8f bwmma(v16bf a, v16bf b, v8f c) {
    return __builtin_amdgcn_wmma_f32_16x16x32_bf16(
        /*neg_a=*/false, a, /*neg_b=*/false, b,
        /*c_mod=*/(short)0, c, /*reuse_a=*/false, /*reuse_b=*/false);
}

// 16-byte copy global -> LDS. Async (ASYNCcnt-tracked) when available.
__device__ inline void copy16_g2l(const __bf16* gsrc, __bf16* ldst) {
#if USE_ASYNC_LDS
    __builtin_amdgcn_global_load_async_to_lds_b128(
        (as1_v4i_p)(as1_void_p)gsrc,
        (as3_v4i_p)(as3_void_p)ldst, 0, 0);
#else
    *reinterpret_cast<v8bf*>(ldst) = *reinterpret_cast<const v8bf*>(gsrc);
#endif
}

__device__ inline void wait_async_all() {
#if USE_ASYNC_LDS
#if __has_builtin(__builtin_amdgcn_s_wait_asynccnt)
    __builtin_amdgcn_s_wait_asynccnt(0);
#else
    asm volatile("s_wait_asynccnt 0x0" ::: "memory");
#endif
#endif
}

// A-matrix fragment, 16x32 bf16 (MxK), src row-major with leading dim `ld`.
// ISA layout: lanes 0-15 hold row m=lane with K {0..7,16..23}; lanes 16-31
// hold row m=lane-16 with K {8..15,24..31}.  Two 16B chunks per lane.
__device__ inline v16bf load_a16(const __bf16* base, int ld, int row, int k0, int lane) {
    const __bf16* r = base + (size_t)row * ld;
    int kg = (lane >> 4) & 1;
    v8bf c0 = *reinterpret_cast<const v8bf*>(r + k0 + kg * 8);
    v8bf c1 = *reinterpret_cast<const v8bf*>(r + k0 + 16 + kg * 8);
    v16bf f;
#pragma unroll
    for (int i = 0; i < 8; ++i) { f[i] = c0[i]; f[8 + i] = c1[i]; }
    return f;
}

// B-matrix fragment, 32x16 bf16 (KxN).  We load B^T rows (row n, K contiguous).
// ISA layout: lanes 0-15 hold col n=lane with K {0..7,16..23}; lanes 16-31 hold K {8..15,24..31}.
__device__ inline v16bf load_b16(const __bf16* base, int ld, int row, int k0, int lane) {
    const __bf16* r = base + (size_t)row * ld;
    int kg = (lane >> 4) & 1;
    v8bf c0 = *reinterpret_cast<const v8bf*>(r + k0 + kg * 8);
    v8bf c1 = *reinterpret_cast<const v8bf*>(r + k0 + 16 + kg * 8);
    v16bf f;
#pragma unroll
    for (int i = 0; i < 8; ++i) { f[i] = c0[i]; f[8 + i] = c1[i]; }
    return f;
}

// ---------------------------------------------------------------------------
// Elementwise fp32 -> bf16 convert.
__global__ void f2bf_kernel(const float* __restrict__ X, __bf16* __restrict__ Y, int n) {
    int i = blockIdx.x * blockDim.x + threadIdx.x;
    if (i < n) Y[i] = (__bf16)X[i];
}

// RoPE: X fp32 (s, nh, 64) -> Y bf16 same layout; `scale` folds 1/sqrt(HD) into Q.
__global__ void rope_kernel(const float* __restrict__ X, const float* __restrict__ fc,
                            __bf16* __restrict__ Y, int nh, float scale) {
    int idx = blockIdx.x * blockDim.x + threadIdx.x;     // over S*nh*32
    int total = S_LEN * nh * 32;
    if (idx >= total) return;
    int i = idx & 31;                    // rotary pair index (HD/2 = 32)
    int h = (idx >> 5) % nh;
    int s = idx / (nh * 32);
    float c  = fc[((size_t)s * 32 + i) * 2 + 0];
    float sn = fc[((size_t)s * 32 + i) * 2 + 1];
    size_t base = ((size_t)s * nh + h) * HEAD_DIM + 2 * i;
    float x0 = X[base], x1 = X[base + 1];
    Y[base]     = (__bf16)((x0 * c - x1 * sn) * scale);
    Y[base + 1] = (__bf16)((x0 * sn + x1 * c) * scale);
}

// V fp32 (s, g, d) -> Vt bf16 (g, d, s)  so P*V operand is K(=t)-contiguous.
__global__ void vtrans_kernel(const float* __restrict__ V, __bf16* __restrict__ Vt) {
    int idx = blockIdx.x * blockDim.x + threadIdx.x;     // over S*512
    if (idx >= S_LEN * N_KV * HEAD_DIM) return;
    int d = idx & 63;
    int g = (idx >> 6) & 7;
    int s = idx >> 9;
    Vt[((size_t)g * HEAD_DIM + d) * S_LEN + s] = (__bf16)V[idx];
}

// ---------------------------------------------------------------------------
// C(MxN,f32) = A(MxK,bf16 rowmajor) * B(NxK,bf16 rowmajor)^T.
// Block = 4 waves, block tile 64(M) x 64(N); each wave owns 16 M rows.
// B tile (64 x 32k, 4 KB) is shared by all 4 waves and double-buffered in LDS
// via async global->LDS copies overlapped with the WMMAs of the current tile.
__global__ __launch_bounds__(128) void gemm_tn_bf16(
        const __bf16* __restrict__ A, const __bf16* __restrict__ B,
        float* __restrict__ C, int K, int lda, int ldb, int ldc) {
    __shared__ alignas(16) __bf16 Blds[2][64][32];
    int tid  = threadIdx.x;
    int lane = tid & 31;
    int wave = tid >> 5;
    int ln = lane & 15, hi = lane >> 4;
    int m0 = blockIdx.y * 64 + wave * 16;
    int n0 = blockIdx.x * 64;

    // Each thread copies two 16B chunks of the 256-chunk (4 KB) B tile.
    int c0 = tid, c1 = tid + 128;
    int r0 = c0 >> 2, o0 = (c0 & 3) * 8;
    int r1 = c1 >> 2, o1 = (c1 & 3) * 8;

    v8f acc0 = {}, acc1 = {}, acc2 = {}, acc3 = {};
    int nK = K / 32;

    // Prologue: stage B tile for k0 = 0 into buffer 0.
    copy16_g2l(B + (size_t)(n0 + r0) * ldb + o0, &Blds[0][r0][o0]);
    copy16_g2l(B + (size_t)(n0 + r1) * ldb + o1, &Blds[0][r1][o1]);

    for (int it = 0; it < nK; ++it) {
        int k0 = it * 32;
        int cur = it & 1;
        wait_async_all();          // my copies for buffer `cur` have landed
        __syncthreads();           // everyone's copies landed; prior reads done
        if (it + 1 < nK) {         // stage next tile into the other buffer
            copy16_g2l(B + (size_t)(n0 + r0) * ldb + (k0 + 32) + o0, &Blds[cur ^ 1][r0][o0]);
            copy16_g2l(B + (size_t)(n0 + r1) * ldb + (k0 + 32) + o1, &Blds[cur ^ 1][r1][o1]);
            __builtin_prefetch(A + (size_t)(m0 + ln) * lda + k0 + 32, 0, 0);
        }
        v16bf a  = load_a16(A, lda, m0 + ln, k0, lane);
        v16bf b0 = load_b16(&Blds[cur][0][0], 32, 0 * 16 + ln, 0, lane);
        v16bf b1 = load_b16(&Blds[cur][0][0], 32, 1 * 16 + ln, 0, lane);
        v16bf b2 = load_b16(&Blds[cur][0][0], 32, 2 * 16 + ln, 0, lane);
        v16bf b3 = load_b16(&Blds[cur][0][0], 32, 3 * 16 + ln, 0, lane);
        acc0 = bwmma(a, b0, acc0);
        acc1 = bwmma(a, b1, acc1);
        acc2 = bwmma(a, b2, acc2);
        acc3 = bwmma(a, b3, acc3);
    }
    // C/D layout: lane ln holds col n; VGPR v holds row m = v + 8*hi.
#pragma unroll
    for (int v = 0; v < 8; ++v) {
        size_t row = (size_t)(m0 + v + 8 * hi) * ldc;
        C[row + n0 + 0 * 16 + ln] = acc0[v];
        C[row + n0 + 1 * 16 + ln] = acc1[v];
        C[row + n0 + 2 * 16 + ln] = acc2[v];
        C[row + n0 + 3 * 16 + ln] = acc3[v];
    }
}

// ---------------------------------------------------------------------------
// Flash attention: one wave per (16 q-rows, head). Q already scaled by 1/8.
// Q: (s, 32, 64) bf16 rope'd; Kb: (t, 8, 64) bf16 rope'd; Vt: (g, 64, t) bf16.
// O: (s, 32, 64) bf16.
__global__ __launch_bounds__(32) void attn_fa_kernel(
        const __bf16* __restrict__ Q, const __bf16* __restrict__ Kb,
        const __bf16* __restrict__ Vt, __bf16* __restrict__ O) {
    int lane = threadIdx.x & 31;
    int ln = lane & 15, hi = lane >> 4;
    int qt = blockIdx.x;               // q tile (16 rows)
    int h  = blockIdx.y;               // 0..31
    int g  = h >> 2;                   // kv head (R=4)
    int qrow0 = qt * 16;

    const __bf16* qbase = Q  + (size_t)h * HEAD_DIM;            // row stride 2048
    const __bf16* kbase = Kb + (size_t)g * HEAD_DIM;            // row stride 512
    const __bf16* vbase = Vt + (size_t)g * HEAD_DIM * S_LEN;    // row stride 2048

    __shared__ alignas(16) __bf16 Plds[16][32];

    float run_max[8], run_sum[8], scl[8];
    v8f oacc[4] = {};
#pragma unroll
    for (int v = 0; v < 8; ++v) { run_max[v] = -3.0e38f; run_sum[v] = 0.0f; }

    // Preload Q fragments (K-dim = 64 -> two 16x32 fragments).
    v16bf qf0 = load_a16(qbase, D_MODEL, qrow0 + ln, 0,  lane);
    v16bf qf1 = load_a16(qbase, D_MODEL, qrow0 + ln, 32, lane);

    int nblk = (qrow0 + 15) / 32 + 1;          // causal: keys up to q row
    for (int blk = 0; blk < nblk; ++blk) {
        int t0 = blk * 32;
        // S = Q * K^T for 32 key columns (two 16-wide fragments).
        v8f s0 = {}, s1 = {};
        {
            v16bf b00 = load_b16(kbase, N_KV * HEAD_DIM, t0 + ln,      0,  lane);
            v16bf b10 = load_b16(kbase, N_KV * HEAD_DIM, t0 + 16 + ln, 0,  lane);
            s0 = bwmma(qf0, b00, s0);
            s1 = bwmma(qf0, b10, s1);
            v16bf b01 = load_b16(kbase, N_KV * HEAD_DIM, t0 + ln,      32, lane);
            v16bf b11 = load_b16(kbase, N_KV * HEAD_DIM, t0 + 16 + ln, 32, lane);
            s0 = bwmma(qf1, b01, s0);
            s1 = bwmma(qf1, b11, s1);
        }
        // Causal mask: only the diagonal block can clip.
        if (t0 + 31 >= qrow0) {
#pragma unroll
            for (int v = 0; v < 8; ++v) {
                int q = qrow0 + v + 8 * hi;
                if (t0 + ln > q)      s0[v] = -1.0e30f;
                if (t0 + 16 + ln > q) s1[v] = -1.0e30f;
            }
        }
        // Online softmax; row m = v + 8*hi, its 16 values live across one half.
#pragma unroll
        for (int v = 0; v < 8; ++v) {
            float m = fmaxf(s0[v], s1[v]);
#pragma unroll
            for (int off = 1; off < 16; off <<= 1)
                m = fmaxf(m, __shfl_xor(m, off, 32));
            float nm = fmaxf(run_max[v], m);
            scl[v] = __expf(run_max[v] - nm);
            run_max[v] = nm;
            float p0 = __expf(s0[v] - nm);
            float p1 = __expf(s1[v] - nm);
            s0[v] = p0; s1[v] = p1;
            float rs = p0 + p1;
#pragma unroll
            for (int off = 1; off < 16; off <<= 1)
                rs += __shfl_xor(rs, off, 32);
            run_sum[v] = run_sum[v] * scl[v] + rs;
        }
        // Stage P (16x32) in LDS to re-fragment into the A layout.
#pragma unroll
        for (int v = 0; v < 8; ++v) {
            int r = v + 8 * hi;
            Plds[r][ln]      = (__bf16)s0[v];
            Plds[r][16 + ln] = (__bf16)s1[v];
        }
        __syncthreads();
        v16bf pf = load_a16(&Plds[0][0], 32, ln, 0, lane);
#pragma unroll
        for (int j = 0; j < 4; ++j) {
#pragma unroll
            for (int v = 0; v < 8; ++v) oacc[j][v] *= scl[v];
            v16bf vf = load_b16(vbase, S_LEN, j * 16 + ln, t0, lane);
            oacc[j] = bwmma(pf, vf, oacc[j]);
        }
        __syncthreads();
    }
    // Normalize and store bf16 (s, h, d).
#pragma unroll
    for (int j = 0; j < 4; ++j) {
#pragma unroll
        for (int v = 0; v < 8; ++v) {
            size_t row = (size_t)(qrow0 + v + 8 * hi) * D_MODEL;
            O[row + h * HEAD_DIM + j * 16 + ln] = (__bf16)(oacc[j][v] / run_sum[v]);
        }
    }
}

// ---------------------------------------------------------------------------
extern "C" void kernel_launch(void* const* d_in, const int* in_sizes, int n_in,
                              void* d_out, int out_size, void* d_ws, size_t ws_size,
                              hipStream_t stream) {
    (void)in_sizes; (void)n_in; (void)out_size; (void)ws_size;
    const float* x  = (const float*)d_in[0];
    const float* fc = (const float*)d_in[1];
    const float* wq = (const float*)d_in[2];
    const float* wk = (const float*)d_in[3];
    const float* wv = (const float*)d_in[4];
    const float* wo = (const float*)d_in[5];
    float* out = (float*)d_out;

    // Workspace carve (256B aligned).
    char* p = (char*)d_ws;
    auto carve = [&](size_t bytes) -> char* {
        char* r = p;
        p += (bytes + 255) & ~(size_t)255;
        return r;
    };
    const size_t SD  = (size_t)S_LEN * D_MODEL;             // 4M
    const size_t SKV = (size_t)S_LEN * N_KV * HEAD_DIM;     // 1M
    __bf16* xb   = (__bf16*)carve(SD * 2);
    __bf16* wqb  = (__bf16*)carve(SD * 2);
    __bf16* wkb  = (__bf16*)carve(SKV * 2);
    __bf16* wvb  = (__bf16*)carve(SKV * 2);
    __bf16* wob  = (__bf16*)carve(SD * 2);
    float*  Qraw = (float*)carve(SD * 4);
    float*  Kraw = (float*)carve(SKV * 4);
    float*  Vraw = (float*)carve(SKV * 4);
    __bf16* Qb   = (__bf16*)carve(SD * 2);
    __bf16* Kb   = (__bf16*)carve(SKV * 2);
    __bf16* Vtb  = (__bf16*)carve(SKV * 2);
    __bf16* AOb  = (__bf16*)carve(SD * 2);

    const int CT = 256;
    // 1) fp32 -> bf16 conversions.
    f2bf_kernel<<<(int)((SD  + CT - 1) / CT), CT, 0, stream>>>(x,  xb,  (int)SD);
    f2bf_kernel<<<(int)((SD  + CT - 1) / CT), CT, 0, stream>>>(wq, wqb, (int)SD);
    f2bf_kernel<<<(int)((SKV + CT - 1) / CT), CT, 0, stream>>>(wk, wkb, (int)SKV);
    f2bf_kernel<<<(int)((SKV + CT - 1) / CT), CT, 0, stream>>>(wv, wvb, (int)SKV);
    f2bf_kernel<<<(int)((SD  + CT - 1) / CT), CT, 0, stream>>>(wo, wob, (int)SD);

    // 2) QKV projections (TN GEMMs), block tile 64x64, 4 waves.
    gemm_tn_bf16<<<dim3(D_MODEL / 64, S_LEN / 64), 128, 0, stream>>>(
        xb, wqb, Qraw, D_MODEL, D_MODEL, D_MODEL, D_MODEL);
    gemm_tn_bf16<<<dim3((N_KV * HEAD_DIM) / 64, S_LEN / 64), 128, 0, stream>>>(
        xb, wkb, Kraw, D_MODEL, D_MODEL, D_MODEL, N_KV * HEAD_DIM);
    gemm_tn_bf16<<<dim3((N_KV * HEAD_DIM) / 64, S_LEN / 64), 128, 0, stream>>>(
        xb, wvb, Vraw, D_MODEL, D_MODEL, D_MODEL, N_KV * HEAD_DIM);

    // 3) RoPE (scale 1/sqrt(64) folded into Q) and V transpose.
    {
        int nq = S_LEN * N_HEADS * 32;
        rope_kernel<<<(nq + CT - 1) / CT, CT, 0, stream>>>(Qraw, fc, Qb, N_HEADS, 0.125f);
        int nk = S_LEN * N_KV * 32;
        rope_kernel<<<(nk + CT - 1) / CT, CT, 0, stream>>>(Kraw, fc, Kb, N_KV, 1.0f);
        int nv = (int)SKV;
        vtrans_kernel<<<(nv + CT - 1) / CT, CT, 0, stream>>>(Vraw, Vtb);
    }

    // 4) Flash attention.
    attn_fa_kernel<<<dim3(S_LEN / 16, N_HEADS), 32, 0, stream>>>(Qb, Kb, Vtb, AOb);

    // 5) Output projection -> fp32 d_out.
    gemm_tn_bf16<<<dim3(D_MODEL / 64, S_LEN / 64), 128, 0, stream>>>(
        AOb, wob, out, D_MODEL, D_MODEL, D_MODEL, D_MODEL);
}
